// OneCrfCCKSDecoder_29197187678475
// MI455X (gfx1250) — matrix operand order repaired and checked by reading, not compiled, not run on hardware
//
#include <hip/hip_runtime.h>
#include <hip/hip_bf16.h>

typedef __attribute__((ext_vector_type(16))) _Float16 v16h;
typedef __attribute__((ext_vector_type(8)))  _Float16 v8h;
typedef __attribute__((ext_vector_type(8)))  float    v8f;
typedef __attribute__((ext_vector_type(4)))  float    v4f;

#define TT    128   // decode length
#define BB    512   // batch
#define EE    128   // entity size
#define EVL   8     // event length
#define NEV   16    // events
#define BTILE 16    // batch rows per block
#define NBLK  (BB / BTILE)  // 32 blocks
#define LDSH  136   // f16 row stride (padded, 16B-aligned rows, conflict-free)
#define LDSF  140   // f32 row stride (padded, 16B-aligned rows)

#define LOG2E 1.4426950408889634f
#define LN2   0.6931471805599453f

// raw hardware transcendentals (v_exp_f32 / v_log_f32 are base-2)
__device__ __forceinline__ float fast_exp(float x) {
  return __builtin_amdgcn_exp2f(x * LOG2E);
}
__device__ __forceinline__ float fast_log(float x) {
  return __builtin_amdgcn_logf(x) * LN2;
}

__global__ __launch_bounds__(256) void crf_fwd_kernel(
    const float* __restrict__ feats,   // [T, B, E]
    const float* __restrict__ trans,   // [E, E]
    float* __restrict__ partials)      // [NBLK]
{
  __shared__ _Float16 TexpS[EE * LDSH];    // exp(trans - rowmax), f16
  __shared__ _Float16 XS[BTILE * LDSH];    // exp(x - rowmax), f16
  __shared__ float    fvS[BTILE * LDSF];   // running fv, f32
  __shared__ float    mtS[EE];
  __shared__ float    mxS[BTILE];
  __shared__ float    bsumS[BTILE];

  const int tid  = threadIdx.x;
  const int lane = tid & 31;
  const int wv   = tid >> 5;        // wave id 0..7 -> j-tile
  const int g    = lane >> 4;       // lane group (wave32 half)
  const int nn   = lane & 15;

  const int bloc = tid >> 4;        // 0..15 local batch row (scalar phases)
  const int ks   = (tid & 15) * 8;  // k slice start (8 elems per thread)
  const int bglb = blockIdx.x * BTILE + bloc;

  // ---- init fv: NEG everywhere, batch row 0 (global) is all-zero ----
  {
    const float v = (bglb == 0) ? 0.0f : -10000.0f;
    for (int i = 0; i < 8; ++i) fvS[bloc * LDSF + ks + i] = v;
  }

  // ---- precompute Texp[j][k] = exp(trans[j][k] - mt[j]) as f16 ----
  {
    const int j = tid >> 1;
    const int h = tid & 1;
    const float* tp = trans + j * EE + h * 64;
    float m = -3.0e38f;
    for (int i = 0; i < 64; i += 4) {
      v4f v = *(const v4f*)(tp + i);
      m = fmaxf(m, fmaxf(fmaxf(v[0], v[1]), fmaxf(v[2], v[3])));
    }
    m = fmaxf(m, __shfl_xor(m, 1, 32));   // pair shares row j
    for (int i = 0; i < 64; i += 8) {
      v4f v0 = *(const v4f*)(tp + i);
      v4f v1 = *(const v4f*)(tp + i + 4);
      v8h o;
      for (int d = 0; d < 4; ++d) {
        o[d]     = (_Float16)fast_exp(v0[d] - m);
        o[d + 4] = (_Float16)fast_exp(v1[d] - m);
      }
      *(v8h*)&TexpS[j * LDSH + h * 64 + i] = o;  // 16B-aligned b128 store
    }
    if (h == 0) mtS[j] = m;
  }

  // preload transitions[-1] slice used by alpha accumulation
  float trr[8];
  for (int i = 0; i < 8; ++i) trr[i] = trans[(EE - 1) * EE + ks + i];

  __syncthreads();

  // ---- constant B-matrix fragments: B[k][n] = Texp[16*wv+n][k] ----
  // f16 B 32x16, contiguous k order chosen to match A: lane group g holds K = Kc + 16g + halfidx, N = lane%16
  v16h bfrag[4];
  for (int c = 0; c < 4; ++c) {
    const _Float16* p = &TexpS[(16 * wv + nn) * LDSH + 32 * c + 16 * g];
    v8h lo = *(const v8h*)p;
    v8h hi = *(const v8h*)(p + 8);
    for (int i = 0; i < 8; ++i) { bfrag[c][i] = lo[i]; bfrag[c][i + 8] = hi[i]; }
  }
  const float mtj = mtS[16 * wv + nn];

  float alphaAcc = 0.0f;

  for (int e = 0; e < NEV; ++e) {
    for (int s = 1; s < EVL; ++s) {
      const int t = e * EVL + s;

      // ---- phase A: x = fv + feat, row max, X = exp(x - mx) -> LDS f16 ----
      const float* fp = feats + ((size_t)t * BB + blockIdx.x * BTILE + bloc) * EE + ks;
      v4f f0 = *(const v4f*)fp;
      v4f f1 = *(const v4f*)(fp + 4);
      __builtin_prefetch(fp + (size_t)BB * EE, 0, 3);  // next t, same slice
      float x[8];
      for (int i = 0; i < 4; ++i) {
        x[i]     = fvS[bloc * LDSF + ks + i]     + f0[i];
        x[i + 4] = fvS[bloc * LDSF + ks + 4 + i] + f1[i];
      }
      float m = x[0];
      for (int i = 1; i < 8; ++i) m = fmaxf(m, x[i]);
      for (int msk = 8; msk >= 1; msk >>= 1) m = fmaxf(m, __shfl_xor(m, msk, 32));
      {
        v8h o;
        for (int i = 0; i < 8; ++i) o[i] = (_Float16)fast_exp(x[i] - m);
        *(v8h*)&XS[bloc * LDSH + ks] = o;  // single b128 store
      }
      if ((tid & 15) == 0) mxS[bloc] = m;
      __syncthreads();

      // ---- phase B: acc[b,j] = sum_k X[b,k] * Texp[j,k] via WMMA ----
      // f16 A 16x32 layout: lane group g holds K = Kc+16g+0..15 (contiguous,
      // matching B), M = lane%16
      // Load all four A fragments into independent registers first so the
      // scheduler can issue all 8 ds_load_b128 back-to-back and overlap the
      // WMMA chain with outstanding LDS loads via partial dscnt waits.
      v16h afrag[4];
      for (int c = 0; c < 4; ++c) {
        const _Float16* pa = &XS[nn * LDSH + 32 * c + 16 * g];
        v8h lo = *(const v8h*)pa;
        v8h hi = *(const v8h*)(pa + 8);
        for (int i = 0; i < 8; ++i) { afrag[c][i] = lo[i]; afrag[c][i + 8] = hi[i]; }
      }
      v8f acc = {};
      for (int c = 0; c < 4; ++c) {
        acc = __builtin_amdgcn_wmma_f32_16x16x32_f16(
            /*neg_a=*/false, afrag[c], /*neg_b=*/false, bfrag[c],
            /*c_mod=*/(short)0, acc, /*reuse_a=*/false, /*reuse_b=*/false);
      }
      // C/D layout: vgpr r, lane group g -> M = r + 8g, N = lane%16
      for (int r = 0; r < 8; ++r) {
        const int br = r + 8 * g;
        const float v = mxS[br] + mtj + fast_log(fmaxf(acc[r], 1e-30f));
        fvS[br * LDSF + 16 * wv + nn] = v;
      }
      __syncthreads();
    }

    // ---- event boundary / terminal: alpha += mean_b lse_k(fv + trans[-1]) ----
    {
      float x[8];
      for (int i = 0; i < 8; ++i) x[i] = fvS[bloc * LDSF + ks + i] + trr[i];
      float m = x[0];
      for (int i = 1; i < 8; ++i) m = fmaxf(m, x[i]);
      for (int msk = 8; msk >= 1; msk >>= 1) m = fmaxf(m, __shfl_xor(m, msk, 32));
      float sum = 0.0f;
      for (int i = 0; i < 8; ++i) sum += fast_exp(x[i] - m);
      for (int msk = 8; msk >= 1; msk >>= 1) sum += __shfl_xor(sum, msk, 32);
      if ((tid & 15) == 0) bsumS[bloc] = m + fast_log(sum);  // sum >= 1, normal range
      __syncthreads();
      float ev = 0.0f;
      for (int q = 0; q < BTILE; ++q) ev += bsumS[q];  // redundant per-thread, deterministic
      alphaAcc += ev;
      __syncthreads();
    }
  }

  if (tid == 0) partials[blockIdx.x] = alphaAcc;
}

__global__ void crf_reduce_kernel(const float* __restrict__ partials,
                                  float* __restrict__ out)
{
  if (threadIdx.x == 0 && blockIdx.x == 0) {
    float s = 0.0f;
    for (int i = 0; i < NBLK; ++i) s += partials[i];   // fixed order -> deterministic
    out[0] = s / ((float)BB * (float)NEV);
  }
}

extern "C" void kernel_launch(void* const* d_in, const int* in_sizes, int n_in,
                              void* d_out, int out_size, void* d_ws, size_t ws_size,
                              hipStream_t stream) {
  const float* feats = (const float*)d_in[0];   // [128, 512, 128] f32
  const float* trans = (const float*)d_in[1];   // [128, 128] f32
  float* out = (float*)d_out;                   // scalar f32
  float* ws  = (float*)d_ws;                    // >= 32 floats

  crf_fwd_kernel<<<dim3(NBLK), dim3(256), 0, stream>>>(feats, trans, ws);
  crf_reduce_kernel<<<dim3(1), dim3(32), 0, stream>>>(ws, out);
}
